// FractalAttention_55052890800918
// MI455X (gfx1250) — matrix operand order repaired and checked by reading, not compiled, not run on hardware
//
#include <hip/hip_runtime.h>

// ---------------------------------------------------------------------------
// FractalAttention for MI455X (gfx1250, wave32, WMMA f32_16x16x32_f16)
//   B=2, T=2048, C=1024, H=16, D=64
//   qkv GEMM -> flash attention -> out GEMM; f16 compute / f32 accumulate.
//   CDNA5 paths: v_wmma_f32_16x16x32_f16, global_load_async_to_lds_b128
//   (ASYNCcnt) for conversion-free LDS staging.
// ---------------------------------------------------------------------------

typedef __attribute__((ext_vector_type(16))) _Float16 v16h;
typedef __attribute__((ext_vector_type(8)))  _Float16 v8h;
typedef __attribute__((ext_vector_type(8)))  float    v8f;
typedef __attribute__((ext_vector_type(4)))  float    f32x4;

union V16U { v16h v; v8h h[2]; };

#define T_SEQ 2048
#define DHEAD 64
#define CEMB  1024

__device__ __forceinline__ v8f wmma32(v16h a, v16h b, v8f c) {
    return __builtin_amdgcn_wmma_f32_16x16x32_f16(
        false, a, false, b, (short)0, c, false, false);
}

// Async 16B global->LDS copy (GV mode), tracked by ASYNCcnt.
__device__ __forceinline__ void async_ld_b128(const void* lds_ptr, const void* gptr) {
    unsigned lds = (unsigned)(uintptr_t)lds_ptr;  // LDS aperture: addr[31:0] = LDS offset
    asm volatile("global_load_async_to_lds_b128 %0, %1, off"
                 :: "v"(lds), "v"((unsigned long long)(uintptr_t)gptr)
                 : "memory");
}
__device__ __forceinline__ void wait_async0() {
    asm volatile("s_wait_asynccnt 0" ::: "memory");
}

// =====================  Kernel 1: QKV projection GEMM  =====================
// out[m,n] = sum_k x[m,k]*qkv_w[n,k] + qkv_b[n];  M=4096, N=3072, K=1024.
// Block tile 128x128, 8 waves (4 row-groups x 2 col-groups); each wave:
// 2 A-frags x 4 B-frags = 8 WMMAs per 32-wide K step.
__global__ __launch_bounds__(256)
void qkv_gemm_kernel(const float* __restrict__ x,
                     const float* __restrict__ w,
                     const float* __restrict__ bias,
                     _Float16* __restrict__ qO,
                     _Float16* __restrict__ kO,
                     _Float16* __restrict__ vO)
{
    __shared__ _Float16 At[128][40];  // 128 rows x 32 k (+8 pad -> 16B aligned rows)
    __shared__ _Float16 Bt[128][40];

    const int tid  = threadIdx.x;
    const int wave = tid >> 5, lane = tid & 31;
    const int nblk = blockIdx.x % 24, mblk = blockIdx.x / 24;
    const int m0 = mblk * 128, n0 = nblk * 128;
    const int waveRow = wave & 3, waveCol = wave >> 2;
    const int ml   = lane & 15;
    const int offA = (lane < 16) ? 0 : 8;    // A-frag interleave select
    const int offB = (lane < 16) ? 0 : 8;    // B-frag interleave select

    v8f acc[2][4];
    for (int rt = 0; rt < 2; ++rt)
        for (int d = 0; d < 4; ++d)
            for (int r = 0; r < 8; ++r) acc[rt][d][r] = 0.0f;

    const int srow = tid >> 1, skp = (tid & 1) * 16;  // staging: 16 elems/thread

    for (int kk = 0; kk < CEMB; kk += 32) {
        { // stage A and B tiles: fp32 -> f16
            const float* asrc = x + (long)(m0 + srow) * CEMB + kk + skp;
            const float* bsrc = w + (long)(n0 + srow) * CEMB + kk + skp;
            for (int c = 0; c < 2; ++c) {
                f32x4 a0 = *(const f32x4*)(asrc + c * 8);
                f32x4 a1 = *(const f32x4*)(asrc + c * 8 + 4);
                f32x4 b0 = *(const f32x4*)(bsrc + c * 8);
                f32x4 b1 = *(const f32x4*)(bsrc + c * 8 + 4);
                v8h ha, hb;
                for (int i = 0; i < 4; ++i) {
                    ha[i] = (_Float16)a0[i]; ha[4 + i] = (_Float16)a1[i];
                    hb[i] = (_Float16)b0[i]; hb[4 + i] = (_Float16)b1[i];
                }
                *(v8h*)(&At[srow][skp + c * 8]) = ha;
                *(v8h*)(&Bt[srow][skp + c * 8]) = hb;
            }
        }
        __syncthreads();

        V16U a[2];
        for (int rt = 0; rt < 2; ++rt) {
            const _Float16* arp = &At[waveRow * 32 + rt * 16 + ml][0];
            a[rt].h[0] = *(const v8h*)(arp + offA);
            a[rt].h[1] = *(const v8h*)(arp + 16 + offA);
        }
        V16U bf[4];
        for (int dt = 0; dt < 4; ++dt) {
            const _Float16* brp = &Bt[waveCol * 64 + dt * 16 + ml][0];
            bf[dt].h[0] = *(const v8h*)(brp + offB);
            bf[dt].h[1] = *(const v8h*)(brp + offB + 16);
        }
        for (int rt = 0; rt < 2; ++rt)
            for (int dt = 0; dt < 4; ++dt)
                acc[rt][dt] = wmma32(a[rt].v, bf[dt].v, acc[rt][dt]);
        __syncthreads();
    }

    // Epilogue: bias + scatter to q/k/v in [B, H, T, D] (f16)
    const int half = (lane < 16) ? 0 : 8;
    for (int dt = 0; dt < 4; ++dt) {
        const int ng = n0 + waveCol * 64 + dt * 16 + ml;   // 0..3071
        const int g = ng >> 10, c = ng & 1023;
        const int hh = c >> 6, dd = c & 63;
        _Float16* dst = (g == 0) ? qO : (g == 1) ? kO : vO;
        const float bv = bias[ng];
        for (int rt = 0; rt < 2; ++rt)
            for (int r = 0; r < 8; ++r) {
                const int mg = m0 + waveRow * 32 + rt * 16 + r + half;
                const int bb = mg >> 11, tt = mg & 2047;
                dst[(((long)bb * 16 + hh) * T_SEQ + tt) * DHEAD + dd] =
                    (_Float16)(acc[rt][dt][r] + bv);
            }
    }
}

// =====================  Kernel 2: Flash attention  =========================
// One wave = 16 query rows of one (b,h); block = 8 waves = 128 query rows.
// Hurst scaling and 1/sqrt(D) folded into Q. K tile staged with async
// global->LDS copies; V staged transposed through VGPRs.
__global__ __launch_bounds__(256)
void attn_kernel(const _Float16* __restrict__ qg,
                 const _Float16* __restrict__ kg,
                 const _Float16* __restrict__ vg,
                 _Float16* __restrict__ og,
                 const float* __restrict__ hurst,
                 const float* __restrict__ hscale)
{
    __shared__ _Float16 Kt[32][72];      // [key][d]  -> QK^T B-frags
    __shared__ _Float16 Vt[64][40];      // [d][key]  -> PV  B-frags
    __shared__ _Float16 Pw[8][16][40];   // per-wave P relayout scratch

    const int tid  = threadIdx.x;
    const int wave = tid >> 5, lane = tid & 31;
    const int bh   = blockIdx.x & 31;    // b*16 + h
    const int qblk = blockIdx.x >> 5;    // 0..15
    const int b    = bh >> 4;
    const long bhT = (long)bh * T_SEQ;

    const float qscale = (1.0f + (hurst[b] - 0.5f) * hscale[0]) * 0.125f;

    const int q0 = qblk * 128 + wave * 16;
    const int ml   = lane & 15;
    const int offA = (lane < 16) ? 0 : 8;
    const int offB = (lane < 16) ? 0 : 8;
    const int half = (lane < 16) ? 0 : 8;

    // Load + pre-scale Q fragments (A layout), contraction = head dim 0..63
    V16U aQ[2];
    {
        const _Float16* qrow = qg + (bhT + q0 + ml) * DHEAD;
        for (int f = 0; f < 2; ++f) {
            v8h lo = *(const v8h*)(qrow + f * 32 + offA);
            v8h hi = *(const v8h*)(qrow + f * 32 + 16 + offA);
            for (int i = 0; i < 8; ++i) {
                lo[i] = (_Float16)((float)lo[i] * qscale);
                hi[i] = (_Float16)((float)hi[i] * qscale);
            }
            aQ[f].h[0] = lo; aQ[f].h[1] = hi;
        }
    }

    v8f acc[4], mrow, lrow;
    for (int r = 0; r < 8; ++r) { mrow[r] = -1e30f; lrow[r] = 0.0f; }
    for (int d = 0; d < 4; ++d)
        for (int r = 0; r < 8; ++r) acc[d][r] = 0.0f;

    const int skey = tid >> 3;        // staging key row 0..31
    const int sdp  = (tid & 7) * 8;   // staging d offset 0..56

    for (int s0 = 0; s0 < T_SEQ; s0 += 32) {
        { // stage K via async LDS copy; V transposed via VGPRs
            async_ld_b128(&Kt[skey][sdp],
                          kg + (bhT + s0 + skey) * DHEAD + sdp);
            v8h vv = *(const v8h*)(vg + (bhT + s0 + skey) * DHEAD + sdp);
            for (int i = 0; i < 8; ++i) Vt[sdp + i][skey] = vv[i];
            wait_async0();
        }
        __syncthreads();

        // ---- scores: S = (alpha*Q) @ K^T (keys 0..15 -> c0, 16..31 -> c1)
        V16U bK0a, bK1a, bK0b, bK1b;
        {
            const _Float16* r0 = &Kt[ml][0];
            const _Float16* r1 = &Kt[16 + ml][0];
            bK0a.h[0] = *(const v8h*)(r0 + offB);       bK0a.h[1] = *(const v8h*)(r0 + offB + 16);
            bK1a.h[0] = *(const v8h*)(r0 + 32 + offB);  bK1a.h[1] = *(const v8h*)(r0 + 32 + offB + 16);
            bK0b.h[0] = *(const v8h*)(r1 + offB);       bK0b.h[1] = *(const v8h*)(r1 + offB + 16);
            bK1b.h[0] = *(const v8h*)(r1 + 32 + offB);  bK1b.h[1] = *(const v8h*)(r1 + 32 + offB + 16);
        }
        v8f c0, c1;
        for (int r = 0; r < 8; ++r) { c0[r] = 0.0f; c1[r] = 0.0f; }
        c0 = wmma32(aQ[0].v, bK0a.v, c0);
        c0 = wmma32(aQ[1].v, bK1a.v, c0);
        c1 = wmma32(aQ[0].v, bK0b.v, c1);
        c1 = wmma32(aQ[1].v, bK1b.v, c1);

        // ---- online softmax (C/D frag: row = vgpr idx within lane-half)
        v8f mnew, corr;
        for (int r = 0; r < 8; ++r) {
            float v = fmaxf(c0[r], c1[r]);
            v = fmaxf(v, __shfl_xor(v, 1));
            v = fmaxf(v, __shfl_xor(v, 2));
            v = fmaxf(v, __shfl_xor(v, 4));
            v = fmaxf(v, __shfl_xor(v, 8));
            mnew[r] = fmaxf(mrow[r], v);
            corr[r] = __expf(mrow[r] - mnew[r]);
            float p0 = __expf(c0[r] - mnew[r]);
            float p1 = __expf(c1[r] - mnew[r]);
            c0[r] = p0; c1[r] = p1;
            float s = p0 + p1;
            s += __shfl_xor(s, 1);
            s += __shfl_xor(s, 2);
            s += __shfl_xor(s, 4);
            s += __shfl_xor(s, 8);
            lrow[r] = lrow[r] * corr[r] + s;
            mrow[r] = mnew[r];
        }
        for (int d = 0; d < 4; ++d)
            for (int r = 0; r < 8; ++r) acc[d][r] *= corr[r];

        // ---- relayout P (C/D frag -> A frag) through per-wave LDS
        for (int r = 0; r < 8; ++r) {
            Pw[wave][r + half][ml]      = (_Float16)c0[r];
            Pw[wave][r + half][16 + ml] = (_Float16)c1[r];
        }
        asm volatile("s_wait_dscnt 0" ::: "memory");
        V16U aP;
        {
            const _Float16* pr = &Pw[wave][ml][0];
            aP.h[0] = *(const v8h*)(pr + offA);
            aP.h[1] = *(const v8h*)(pr + 16 + offA);
        }

        // ---- acc += P @ V (batch B-frag loads, then 4 WMMAs)
        V16U bV[4];
        for (int dt = 0; dt < 4; ++dt) {
            const _Float16* vr = &Vt[dt * 16 + ml][0];
            bV[dt].h[0] = *(const v8h*)(vr + offB);
            bV[dt].h[1] = *(const v8h*)(vr + offB + 16);
        }
        for (int dt = 0; dt < 4; ++dt)
            acc[dt] = wmma32(aP.v, bV[dt].v, acc[dt]);
        __syncthreads();
    }

    // Epilogue: normalize rows, store attn to [B, T, C] f16 (C = h*64 + d)
    for (int r = 0; r < 8; ++r) lrow[r] = 1.0f / lrow[r];
    const int hcol = (bh & 15) * DHEAD;
    for (int dt = 0; dt < 4; ++dt)
        for (int r = 0; r < 8; ++r) {
            const int tt = q0 + r + half;
            og[((long)b * T_SEQ + tt) * CEMB + hcol + dt * 16 + ml] =
                (_Float16)(acc[dt][r] * lrow[r]);
        }
}

// =====================  Kernel 3: output projection GEMM  ==================
// out[m,n] = sum_k attn[m,k]*out_w[n,k] + out_b[n];  M=4096, N=K=1024.
// Block tile 128x128; A (f16) staged via async LDS copies, B converted.
__global__ __launch_bounds__(256)
void out_gemm_kernel(const _Float16* __restrict__ a_in,
                     const float* __restrict__ w,
                     const float* __restrict__ bias,
                     float* __restrict__ out)
{
    __shared__ _Float16 At[128][40];
    __shared__ _Float16 Bt[128][40];

    const int tid  = threadIdx.x;
    const int wave = tid >> 5, lane = tid & 31;
    const int nblk = blockIdx.x % 8, mblk = blockIdx.x / 8;
    const int m0 = mblk * 128, n0 = nblk * 128;
    const int waveRow = wave & 3, waveCol = wave >> 2;
    const int ml   = lane & 15;
    const int offA = (lane < 16) ? 0 : 8;
    const int offB = (lane < 16) ? 0 : 8;

    v8f acc[2][4];
    for (int rt = 0; rt < 2; ++rt)
        for (int d = 0; d < 4; ++d)
            for (int r = 0; r < 8; ++r) acc[rt][d][r] = 0.0f;

    const int srow = tid >> 1, skp = (tid & 1) * 16;

    for (int kk = 0; kk < CEMB; kk += 32) {
        // stage A (already f16): two async 16B copies per thread
        {
            const _Float16* asrc = a_in + (long)(m0 + srow) * CEMB + kk + skp;
            async_ld_b128(&At[srow][skp],     asrc);
            async_ld_b128(&At[srow][skp + 8], asrc + 8);
        }
        { // stage B (fp32 weights -> f16)
            const float* bsrc = w + (long)(n0 + srow) * CEMB + kk + skp;
            for (int c = 0; c < 2; ++c) {
                f32x4 b0 = *(const f32x4*)(bsrc + c * 8);
                f32x4 b1 = *(const f32x4*)(bsrc + c * 8 + 4);
                v8h hb;
                for (int i = 0; i < 4; ++i) { hb[i] = (_Float16)b0[i]; hb[4 + i] = (_Float16)b1[i]; }
                *(v8h*)(&Bt[srow][skp + c * 8]) = hb;
            }
        }
        wait_async0();
        __syncthreads();

        V16U a[2];
        for (int rt = 0; rt < 2; ++rt) {
            const _Float16* arp = &At[waveRow * 32 + rt * 16 + ml][0];
            a[rt].h[0] = *(const v8h*)(arp + offA);
            a[rt].h[1] = *(const v8h*)(arp + 16 + offA);
        }
        V16U bf[4];
        for (int dt = 0; dt < 4; ++dt) {
            const _Float16* brp = &Bt[waveCol * 64 + dt * 16 + ml][0];
            bf[dt].h[0] = *(const v8h*)(brp + offB);
            bf[dt].h[1] = *(const v8h*)(brp + offB + 16);
        }
        for (int rt = 0; rt < 2; ++rt)
            for (int dt = 0; dt < 4; ++dt)
                acc[rt][dt] = wmma32(a[rt].v, bf[dt].v, acc[rt][dt]);
        __syncthreads();
    }

    const int half = (lane < 16) ? 0 : 8;
    for (int dt = 0; dt < 4; ++dt) {
        const int ng = n0 + waveCol * 64 + dt * 16 + ml;
        const float bv = bias[ng];
        for (int rt = 0; rt < 2; ++rt)
            for (int r = 0; r < 8; ++r) {
                const int mg = m0 + waveRow * 32 + rt * 16 + r + half;
                out[(long)mg * CEMB + ng] = acc[rt][dt][r] + bv;
            }
    }
}

// ===========================================================================
extern "C" void kernel_launch(void* const* d_in, const int* in_sizes, int n_in,
                              void* d_out, int out_size, void* d_ws, size_t ws_size,
                              hipStream_t stream)
{
    const float* x      = (const float*)d_in[0];
    const float* hurst  = (const float*)d_in[1];
    const float* qkv_w  = (const float*)d_in[2];
    const float* qkv_b  = (const float*)d_in[3];
    const float* out_w  = (const float*)d_in[4];
    const float* out_b  = (const float*)d_in[5];
    const float* hscale = (const float*)d_in[6];
    float* out = (float*)d_out;

    // Workspace (f16): q | k | v | attn = 4 * 8 MB = 32 MB
    const size_t HEAD_ELEMS = (size_t)2 * 16 * T_SEQ * DHEAD;
    _Float16* qf   = (_Float16*)d_ws;
    _Float16* kf   = qf + HEAD_ELEMS;
    _Float16* vf   = kf + HEAD_ELEMS;
    _Float16* attn = vf + HEAD_ELEMS;

    // QKV projection: (4096/128) * (3072/128) = 32 * 24 blocks
    qkv_gemm_kernel<<<32 * 24, 256, 0, stream>>>(x, qkv_w, qkv_b, qf, kf, vf);
    // Attention: (T/128) q-blocks * (B*H) = 16 * 32 = 512 blocks
    attn_kernel<<<512, 256, 0, stream>>>(qf, kf, vf, attn, hurst, hscale);
    // Output projection: (4096/128) * (1024/128) = 32 * 8 blocks
    out_gemm_kernel<<<32 * 8, 256, 0, stream>>>(attn, out_w, out_b, out);
}
